// MyModel_87454124082069
// MI455X (gfx1250) — matrix operand order repaired and checked by reading, not compiled, not run on hardware
//
#include <hip/hip_runtime.h>
#include <hip/hip_bf16.h>

#define T_STEPS 512
#define FEAT    128
#define UNITS   64
#define G3      192          // 3*UNITS
#define NCLS    10
#define BLK     384          // 12 waves * 32 lanes (wave32)
#define ROWS    16           // batch rows per workgroup (WMMA M)
#define NV4     (ROWS * FEAT / 4)   // 512 float4 slots per x-tile

typedef __attribute__((ext_vector_type(16))) __bf16         v16bf;
typedef __attribute__((ext_vector_type(8)))  float          v8f;
typedef __attribute__((ext_vector_type(8)))  unsigned short us8;

__device__ __forceinline__ unsigned short f2bf(float f) {
  // round-to-nearest-even f32 -> bf16
  unsigned int u = __builtin_bit_cast(unsigned int, f);
  u += 0x7FFFu + ((u >> 16) & 1u);
  return (unsigned short)(u >> 16);
}

__device__ __forceinline__ unsigned int pk2bf(float a, float b) {
  return (unsigned int)f2bf(a) | ((unsigned int)f2bf(b) << 16);
}

__device__ __forceinline__ float sigmoidf(float x) {
  return 1.0f / (1.0f + __expf(-x));
}

// Workgroup barrier that fences ONLY LDS (dscnt), not in-flight global loads.
// __syncthreads() would emit s_wait_loadcnt_dscnt 0, stalling the software-
// pipelined global loads every step; all cross-phase data here lives in LDS,
// so a dscnt-only release before the split barrier is sufficient.
__device__ __forceinline__ void lds_barrier() {
  asm volatile("s_wait_dscnt 0x0\n\t"
               "s_barrier_signal -1\n\t"
               "s_barrier_wait -1" ::: "memory");
}

union AV { v16bf v; us8 h[2]; unsigned short s[16]; };

// Fused GRU: each workgroup owns 16 batch rows for all 512 timesteps.
// Wave n (0..11) owns gate-column tile [16n,16n+16) of the 192 gate columns.
// Inputs for step t+2 are software-pipelined in registers (issued in region C
// of step t, committed to LDS in region C of step t+1).
__global__ __launch_bounds__(BLK, 1)
void gru_fused_kernel(const float* __restrict__ inp,   // [512,512,128]
                      const float* __restrict__ W,     // [128,192]
                      const float* __restrict__ U,     // [64,192]
                      const float* __restrict__ b,     // [2,192]
                      const float* __restrict__ W1,    // [64,64]
                      const float* __restrict__ b1,    // [64]
                      const float* __restrict__ W2,    // [64,10]
                      const float* __restrict__ b2,    // [10]
                      float* __restrict__ out) {       // [512,10]
  __shared__ __align__(16) unsigned short sX[ROWS][136];  // x_t tile, bf16 (padded)
  __shared__ __align__(16) unsigned short sH[ROWS][72];   // h, bf16 (padded)
  __shared__ float sHf[ROWS][68];                         // h, f32
  __shared__ float sZR[ROWS][132];                        // z/r (post-sigmoid)
  __shared__ float sXH[ROWS][68];                         // x-part of candidate
  __shared__ float sRH[ROWS][68];                         // rec-part of candidate

  const int tid   = threadIdx.x;
  const int wv    = tid >> 5;       // wave id 0..11  -> gate-column tile
  const int lane  = tid & 31;
  const int col   = lane & 15;      // N-within-tile / A-matrix row
  const int khalf = lane >> 4;      // lane-half selects K sub-block
  const int b0    = blockIdx.x * ROWS;

  // Pipelined input slots: thread owns float4 slots idx0 (always) and idx1
  // (threads 0..127 only).  slot idx -> row m = idx>>5, col k4 = (idx&31)*4.
  const int idx0 = tid;
  const int idx1 = tid + BLK;
  const int m0 = idx0 >> 5, k0 = (idx0 & 31) * 4;
  const int m1 = idx1 >> 5, k1 = (idx1 & 31) * 4;
  const bool has1 = (idx1 < NV4);
  const float* rowp0 = inp + (size_t)(b0 + m0) * T_STEPS * FEAT + k0;
  const float* rowp1 = inp + (size_t)(b0 + m1) * T_STEPS * FEAT + k1;

  // ---- one-time: load W,U column tiles into registers in B-matrix layout ----
  // B 32x16 bf16, this code's contiguous K order (same as A): lanes 0-15 hold K=0..15, lanes 16-31 hold K=16..31,
  // two consecutive K per VGPR.
  const int ncol = wv * 16 + col;
  const int kof  = khalf * 16;
  v16bf bw[4], bu[2];
  for (int c = 0; c < 4; ++c) {
    AV u;
    for (int j = 0; j < 16; ++j)
      u.s[j] = f2bf(W[(size_t)(32 * c + kof + j) * G3 + ncol]);
    bw[c] = u.v;
  }
  for (int c = 0; c < 2; ++c) {
    AV u;
    for (int j = 0; j < 16; ++j)
      u.s[j] = f2bf(U[(size_t)(32 * c + kof + j) * G3 + ncol]);
    bu[c] = u.v;
  }
  const float biasx = b[ncol];        // b[0]
  const float biasr = b[G3 + ncol];   // b[1]

  // ---- h0 = 0 ----
  for (int idx = tid; idx < ROWS * UNITS; idx += BLK) {
    int m = idx >> 6, u = idx & 63;
    sHf[m][u] = 0.0f;
    sH[m][u]  = 0;
  }

  // ---- prologue: stage x-tile for t=0 into LDS, issue loads for t=1 ----
  float4 r0, r1;
  r0 = *(const float4*)(rowp0);
  if (has1) r1 = *(const float4*)(rowp1);
  {
    uint2 p;
    p.x = pk2bf(r0.x, r0.y);
    p.y = pk2bf(r0.z, r0.w);
    *(uint2*)&sX[m0][k0] = p;
    if (has1) {
      p.x = pk2bf(r1.x, r1.y);
      p.y = pk2bf(r1.z, r1.w);
      *(uint2*)&sX[m1][k1] = p;
    }
  }
  r0 = *(const float4*)(rowp0 + (size_t)1 * FEAT);
  if (has1) r1 = *(const float4*)(rowp1 + (size_t)1 * FEAT);
  lds_barrier();

  for (int t = 0; t < T_STEPS; ++t) {
    // ---- Region B: WMMA. xp = x_t@W + b0 (K=128), rec = h@U + b1 (K=64) ----
    {
      v8f ax, ar;
      for (int g = 0; g < 8; ++g) { ax[g] = biasx; ar[g] = biasr; }
      const int ah = khalf * 16;  // A layout: lanes16-31 take K=16..31 (contiguous)
      for (int c = 0; c < 4; ++c) {
        AV a;
        a.h[0] = *(const us8*)&sX[col][32 * c + ah];        // VGPR0-3: K ah..ah+7
        a.h[1] = *(const us8*)&sX[col][32 * c + ah + 8];    // VGPR4-7: K ah+8..ah+15
        ax = __builtin_amdgcn_wmma_f32_16x16x32_bf16(
                 false, a.v, false, bw[c], (short)0, ax, false, false);
      }
      for (int c = 0; c < 2; ++c) {
        AV a;
        a.h[0] = *(const us8*)&sH[col][32 * c + ah];
        a.h[1] = *(const us8*)&sH[col][32 * c + ah + 8];
        ar = __builtin_amdgcn_wmma_f32_16x16x32_bf16(
                 false, a.v, false, bu[c], (short)0, ar, false, false);
      }
      // C/D layout: VGPR g holds row g (lanes0-15) / g+8 (lanes16-31), col = lane&15
      const int rbase = khalf * 8;
      if (wv < 8) {               // z (waves 0-3) and r (waves 4-7): activate here
        const int cc = wv * 16 + col;
        for (int g = 0; g < 8; ++g)
          sZR[rbase + g][cc] = sigmoidf(ax[g] + ar[g]);
      } else {                    // candidate gate: keep xp / rec separate
        const int cc = (wv - 8) * 16 + col;
        for (int g = 0; g < 8; ++g) {
          sXH[rbase + g][cc] = ax[g];
          sRH[rbase + g][cc] = ar[g];
        }
      }
    }
    lds_barrier();

    // ---- Region C: gate blend -> h(t+1); commit x-tile(t+1); issue loads t+2 ----
    for (int idx = tid; idx < ROWS * UNITS; idx += BLK) {
      int m = idx >> 6, u = idx & 63;
      float z  = sZR[m][u];
      float r  = sZR[m][UNITS + u];
      float hh = fmaxf(0.0f, sXH[m][u] + r * sRH[m][u]);
      float hn = z * sHf[m][u] + (1.0f - z) * hh;
      sHf[m][u] = hn;
      sH[m][u]  = f2bf(hn);
    }
    if (t + 1 < T_STEPS) {
      uint2 p;
      p.x = pk2bf(r0.x, r0.y);
      p.y = pk2bf(r0.z, r0.w);
      *(uint2*)&sX[m0][k0] = p;                 // ds_store_b64
      if (has1) {
        p.x = pk2bf(r1.x, r1.y);
        p.y = pk2bf(r1.z, r1.w);
        *(uint2*)&sX[m1][k1] = p;
      }
      if (t + 2 < T_STEPS) {                    // ~1.5 steps of load slack
        r0 = *(const float4*)(rowp0 + (size_t)(t + 2) * FEAT);
        if (has1) r1 = *(const float4*)(rowp1 + (size_t)(t + 2) * FEAT);
      }
      if ((t + 3 < T_STEPS) && ((idx0 & 7) == 0))   // pre-stage GL2 for t+3
        __builtin_prefetch(rowp0 + (size_t)(t + 3) * FEAT, 0, 1);
    }
    lds_barrier();
  }

  // ---- MLP head (tiny): x1 = relu(h @ W1 + b1) ----
  for (int idx = tid; idx < ROWS * 64; idx += BLK) {
    int m = idx >> 6, u = idx & 63;
    float acc = b1[u];
    for (int k = 0; k < UNITS; ++k)
      acc += sHf[m][k] * W1[k * 64 + u];
    sXH[m][u] = fmaxf(acc, 0.0f);
  }
  lds_barrier();
  // logits = x1 @ W2 + b2
  for (int idx = tid; idx < ROWS * NCLS; idx += BLK) {
    int m = idx / NCLS, c = idx % NCLS;
    float acc = b2[c];
    for (int k = 0; k < 64; ++k)
      acc += sXH[m][k] * W2[k * NCLS + c];
    sRH[m][c] = acc;
  }
  lds_barrier();
  // softmax per row
  if (tid < ROWS) {
    int m = tid;
    float mx = -3.0e38f;
    for (int c = 0; c < NCLS; ++c) mx = fmaxf(mx, sRH[m][c]);
    float e[NCLS], s = 0.0f;
    for (int c = 0; c < NCLS; ++c) { e[c] = __expf(sRH[m][c] - mx); s += e[c]; }
    float inv = 1.0f / s;
    for (int c = 0; c < NCLS; ++c)
      out[(size_t)(b0 + m) * NCLS + c] = e[c] * inv;
  }
}

extern "C" void kernel_launch(void* const* d_in, const int* in_sizes, int n_in,
                              void* d_out, int out_size, void* d_ws, size_t ws_size,
                              hipStream_t stream) {
  (void)in_sizes; (void)n_in; (void)out_size; (void)d_ws; (void)ws_size;
  const float* inp = (const float*)d_in[0];
  const float* W   = (const float*)d_in[1];
  const float* U   = (const float*)d_in[2];
  const float* b   = (const float*)d_in[3];
  const float* W1  = (const float*)d_in[4];
  const float* b1  = (const float*)d_in[5];
  const float* W2  = (const float*)d_in[6];
  const float* b2  = (const float*)d_in[7];
  float* out = (float*)d_out;

  gru_fused_kernel<<<512 / ROWS, BLK, 0, stream>>>(inp, W, U, b, W1, b1, W2, b2, out);
}
